// BlockSpaAttn_29850022707949
// MI455X (gfx1250) — matrix operand order repaired and checked by reading, not compiled, not run on hardware
//
#include <hip/hip_runtime.h>
#include <hip/hip_bf16.h>

typedef __bf16 bf16_t;
typedef __attribute__((ext_vector_type(16))) __bf16 v16bf;
typedef __attribute__((ext_vector_type(8)))  __bf16 v8bf;
typedef __attribute__((ext_vector_type(8)))  float  v8f;
typedef __attribute__((ext_vector_type(4)))  float  v4f;

// D = A(16x32 bf16) * B(32x16 bf16) + C(16x16 f32)
__device__ __forceinline__ v8f wmma_bf16(v16bf a, v16bf b, v8f c) {
  return __builtin_amdgcn_wmma_f32_16x16x32_bf16(false, a, false, b, (short)0, c,
                                                 false, false);
}

// A fragment (16x32): lane<16 -> rows m0+lane, K in {k0..k0+15};
// lane>=16 -> same rows, K in {k0+16..k0+31} (matches load_B_tr k order).
__device__ __forceinline__ v16bf load_A(const bf16_t* src, int ld, int m0, int k0,
                                        int lane) {
  int r  = m0 + (lane & 15);
  int kb = k0 + ((lane >> 4) << 4);
  const bf16_t* p = src + r * ld + kb;
  v8bf lo = *(const v8bf*)(p);
  v8bf hi = *(const v8bf*)(p + 8);
  return __builtin_shufflevector(lo, hi, 0,1,2,3,4,5,6,7,8,9,10,11,12,13,14,15);
}

// B fragment (32x16), logical B[k][n] stored TRANSPOSED at src[n*ld + k]
// (used for W^T, K^T, and transposed-V): lane -> column n = n0+lane%16,
// 16 consecutive k -> two 16B LDS loads.
__device__ __forceinline__ v16bf load_B_tr(const bf16_t* src, int ld, int k0, int n0,
                                           int lane) {
  int n  = n0 + (lane & 15);
  int kb = k0 + ((lane >> 4) << 4);
  const bf16_t* p = src + n * ld + kb;
  v8bf lo = *(const v8bf*)(p);
  v8bf hi = *(const v8bf*)(p + 8);
  return __builtin_shufflevector(lo, hi, 0,1,2,3,4,5,6,7,8,9,10,11,12,13,14,15);
}

// Store f32 accumulator tile as bf16 row-major [m][n] (scalar, ISA C/D layout).
__device__ __forceinline__ void store_tile(bf16_t* dst, int ld, int m0, int n0,
                                           v8f acc, float scale, int lane) {
  int n  = n0 + (lane & 15);
  int mb = m0 + ((lane >> 4) << 3);
#pragma unroll
  for (int r = 0; r < 8; ++r) dst[(mb + r) * ld + n] = (bf16_t)(acc[r] * scale);
}

// Store f32 accumulator tile as bf16 TRANSPOSED [n][m]: per lane the 8 rows are
// contiguous -> single 16B ds_store_b128.
__device__ __forceinline__ void store_tile_T(bf16_t* dst, int ld, int m0, int n0,
                                             v8f acc, int lane) {
  int n  = n0 + (lane & 15);
  int mb = m0 + ((lane >> 4) << 3);
  v8bf pk;
#pragma unroll
  for (int r = 0; r < 8; ++r) pk[r] = (bf16_t)acc[r];
  *(v8bf*)(dst + n * ld + mb) = pk;
}

// One workgroup (4 waves) per 8x8 window.  Grid = 8 batches * 64*64 windows.
__global__ __launch_bounds__(128) void swin_window_attn(
    const float* __restrict__ x, const float* __restrict__ wqkv,
    const float* __restrict__ wout, const float* __restrict__ bout,
    float* __restrict__ out) {
  __shared__ __align__(16) char smem[65536];
  bf16_t* s_wqkv = (bf16_t*)(smem);          // 192*64*2 = 24576 B
  float*  s_y    = (float*)(smem);           // 64*64*4  = 16384 B (aliases wqkv after barrier 2), [ch][token]
  bf16_t* s_wout = (bf16_t*)(smem + 24576);  // 8192 B
  bf16_t* s_x    = (bf16_t*)(smem + 32768);  // 8192 B  [token][ch]
  bf16_t* s_k    = (bf16_t*)(smem + 40960);  // 8192 B  [token][ch]
  bf16_t* s_vt   = (bf16_t*)(smem + 49152);  // 8192 B  V TRANSPOSED [ch][token]
  bf16_t* s_qa   = (bf16_t*)(smem + 57344);  // 8192 B  q / attn / attn*v strips

  const int tid  = threadIdx.x;
  const int lane = tid & 31;
  const int wave = tid >> 5;
  const int wid  = blockIdx.x;
  const int batch = wid >> 12;
  const int wy = (wid >> 6) & 63, wx = wid & 63;
  const int h0 = wy * 8, w0 = wx * 8;
  const bool wrapw = (wx == 0);  // only leftmost window column wraps in W

  // Stage weights -> LDS bf16.
  for (int i = tid; i < 192 * 64; i += 128) s_wqkv[i] = (bf16_t)wqkv[i];
  for (int i = tid; i < 64 * 64;  i += 128) s_wout[i] = (bf16_t)wout[i];

  // Gather the shifted window (roll +4,+4 folded into indices).  Unit = one
  // (channel, window-row): 8 contiguous pixels -> two aligned float4 loads.
#pragma unroll
  for (int it = 0; it < 4; ++it) {
    int u = it * 128 + tid;            // 0..511 = 64 ch x 8 rows
    int c = u >> 3, iy = u & 7;
    int sh = (h0 + iy + 508) & 511;
    const float* src = x + ((size_t)(batch * 64 + c) << 18) + ((size_t)sh << 9);
    int t0 = iy * 8;
    if (!wrapw) {
      v4f a = *(const v4f*)(src + w0 - 4);   // (8*wx-4)*4B is 16B aligned
      v4f b = *(const v4f*)(src + w0);
#pragma unroll
      for (int j = 0; j < 4; ++j) {
        s_x[(t0 + j) * 64 + c]     = (bf16_t)a[j];
        s_x[(t0 + 4 + j) * 64 + c] = (bf16_t)b[j];
      }
    } else {
#pragma unroll
      for (int ix = 0; ix < 8; ++ix) {
        int sw = (w0 + ix + 508) & 511;
        s_x[(t0 + ix) * 64 + c] = (bf16_t)src[sw];
      }
    }
  }
  __syncthreads();

  const int m0 = wave * 16;  // this wave's token strip

  // ---- Stage 2: Q,K,V strips = X * W^T ------------------------------------
  v16bf ax0 = load_A(s_x, 64, m0, 0, lane);
  v16bf ax1 = load_A(s_x, 64, m0, 32, lane);
#pragma unroll
  for (int nt = 0; nt < 4; ++nt) {
    v8f aq = {}, ak = {}, av = {};
    aq = wmma_bf16(ax0, load_B_tr(s_wqkv,            64, 0,  nt * 16, lane), aq);
    aq = wmma_bf16(ax1, load_B_tr(s_wqkv,            64, 32, nt * 16, lane), aq);
    ak = wmma_bf16(ax0, load_B_tr(s_wqkv + 64 * 64,  64, 0,  nt * 16, lane), ak);
    ak = wmma_bf16(ax1, load_B_tr(s_wqkv + 64 * 64,  64, 32, nt * 16, lane), ak);
    av = wmma_bf16(ax0, load_B_tr(s_wqkv + 128 * 64, 64, 0,  nt * 16, lane), av);
    av = wmma_bf16(ax1, load_B_tr(s_wqkv + 128 * 64, 64, 32, nt * 16, lane), av);
    store_tile(s_qa, 64, m0, nt * 16, aq, 0.125f, lane);  // q * DIM^-0.5
    store_tile(s_k,  64, m0, nt * 16, ak, 1.0f,   lane);
    store_tile_T(s_vt, 64, m0, nt * 16, av, lane);        // V^T: 1x b128 store
  }
  __syncthreads();  // K,V complete; w_qkv region becomes free (s_y alias)

  // ---- Stage 3: sim = q * k^T --------------------------------------------
  v16bf aq0 = load_A(s_qa, 64, m0, 0, lane);
  v16bf aq1 = load_A(s_qa, 64, m0, 32, lane);
  v8f sim[4];
#pragma unroll
  for (int nt = 0; nt < 4; ++nt) {
    v8f acc = {};
    acc = wmma_bf16(aq0, load_B_tr(s_k, 64, 0,  nt * 16, lane), acc);
    acc = wmma_bf16(aq1, load_B_tr(s_k, 64, 32, nt * 16, lane), acc);
    sim[nt] = acc;
  }

  // ---- Softmax over each row (64 cols = 4 tiles x 16 lanes per half) ------
#pragma unroll
  for (int r = 0; r < 8; ++r) {
    float m = fmaxf(fmaxf(sim[0][r], sim[1][r]), fmaxf(sim[2][r], sim[3][r]));
#pragma unroll
    for (int d = 1; d < 16; d <<= 1) m = fmaxf(m, __shfl_xor(m, d, 16));
    float s = 0.f;
#pragma unroll
    for (int nt = 0; nt < 4; ++nt) {
      float e = __expf(sim[nt][r] - m);
      sim[nt][r] = e;
      s += e;
    }
#pragma unroll
    for (int d = 1; d < 16; d <<= 1) s += __shfl_xor(s, d, 16);
    float inv = 1.f / s;
#pragma unroll
    for (int nt = 0; nt < 4; ++nt) sim[nt][r] *= inv;
  }
#pragma unroll
  for (int nt = 0; nt < 4; ++nt)
    store_tile(s_qa, 64, m0, nt * 16, sim[nt], 1.0f, lane);

  // ---- Stage 4: out = attn * V  (V^T in LDS -> vectorized B loads) --------
  v16bf aa0 = load_A(s_qa, 64, m0, 0, lane);
  v16bf aa1 = load_A(s_qa, 64, m0, 32, lane);
#pragma unroll
  for (int nt = 0; nt < 4; ++nt) {
    v8f acc = {};
    acc = wmma_bf16(aa0, load_B_tr(s_vt, 64, 0,  nt * 16, lane), acc);
    acc = wmma_bf16(aa1, load_B_tr(s_vt, 64, 32, nt * 16, lane), acc);
    store_tile(s_qa, 64, m0, nt * 16, acc, 1.0f, lane);  // reuse strip as next A
  }

  // ---- Stage 5: y = out * Wout^T + bias; stage transposed [ch][token] -----
  v16bf ao0 = load_A(s_qa, 64, m0, 0, lane);
  v16bf ao1 = load_A(s_qa, 64, m0, 32, lane);
#pragma unroll
  for (int nt = 0; nt < 4; ++nt) {
    v8f acc = {};
    acc = wmma_bf16(ao0, load_B_tr(s_wout, 64, 0,  nt * 16, lane), acc);
    acc = wmma_bf16(ao1, load_B_tr(s_wout, 64, 32, nt * 16, lane), acc);
    int n = nt * 16 + (lane & 15);
    float bias = bout[n];
    int mb = m0 + ((lane >> 4) << 3);
    v4f y0, y1;
#pragma unroll
    for (int r = 0; r < 4; ++r) { y0[r] = acc[r] + bias; y1[r] = acc[r + 4] + bias; }
    *(v4f*)(s_y + n * 64 + mb)     = y0;   // contiguous 16B LDS stores
    *(v4f*)(s_y + n * 64 + mb + 4) = y1;
  }

  // ---- Scatter back with inverse roll; float4 global stores off fast path --
  // Each wave handles only its own 16-token strip -> no extra barrier needed.
#pragma unroll
  for (int it = 0; it < 4; ++it) {
    int u = it * 32 + lane;       // 128 units: 64 channels x 2 sub-rows
    int c = u >> 1, sub = u & 1;
    int iy = 2 * wave + sub;
    int dh = (h0 + iy + 508) & 511;
    float* dst = out + ((size_t)(batch * 64 + c) << 18) + ((size_t)dh << 9);
    const float* row = s_y + c * 64 + m0 + sub * 8;  // 8 contiguous f32
    if (!wrapw) {
      v4f a = *(const v4f*)(row);
      v4f b = *(const v4f*)(row + 4);
      *(v4f*)(dst + w0 - 4) = a;   // 16B aligned (32*wx-16 bytes)
      *(v4f*)(dst + w0)     = b;
    } else {
#pragma unroll
      for (int ix = 0; ix < 8; ++ix)
        dst[(w0 + ix + 508) & 511] = row[ix];
    }
  }
}

extern "C" void kernel_launch(void* const* d_in, const int* in_sizes, int n_in,
                              void* d_out, int out_size, void* d_ws, size_t ws_size,
                              hipStream_t stream) {
  const float* x    = (const float*)d_in[0];
  const float* wqkv = (const float*)d_in[1];
  const float* wout = (const float*)d_in[2];
  const float* bout = (const float*)d_in[3];
  float* out = (float*)d_out;
  (void)in_sizes; (void)n_in; (void)out_size; (void)d_ws; (void)ws_size;
  dim3 grid(8 * 64 * 64);   // batch * 64x64 windows
  swin_window_attn<<<grid, 128, 0, stream>>>(x, wqkv, wout, bout, out);
}
